// MultiHeadedSelfAttention_8632884265352
// MI455X (gfx1250) — matrix operand order repaired and checked by reading, not compiled, not run on hardware
//
#include <hip/hip_runtime.h>

// ---------- types ----------
typedef __attribute__((ext_vector_type(16))) __bf16 v16bf;
typedef __attribute__((ext_vector_type(8)))  __bf16 v8bf;
typedef __attribute__((ext_vector_type(8)))  float  v8f;
typedef int v4i __attribute__((vector_size(16)));          // matches builtin param
typedef __attribute__((address_space(1))) v4i* gas_v4i_p;  // global addrspace
typedef __attribute__((address_space(3))) v4i* las_v4i_p;  // LDS addrspace

#define HID   1024
#define NH    16
#define HD    64
#define SEQ   2048
#define BATCH 4
#define SCALE 0.125f   // 1/sqrt(64)

#if __has_builtin(__builtin_amdgcn_global_load_async_to_lds_b128)
#define HAVE_ASYNC 1
#else
#define HAVE_ASYNC 0
#endif

__device__ __forceinline__ __bf16 f2bf(float f) {
  union { float f; unsigned u; } x; x.f = f;
  unsigned r = x.u + 0x7FFFu + ((x.u >> 16) & 1u);   // round-to-nearest-even
  unsigned short h = (unsigned short)(r >> 16);
  union { unsigned short s; __bf16 b; } y; y.s = h;
  return y.b;
}

// 16B cache->LDS copy: async (ASYNCcnt) when the toolchain exposes it.
__device__ __forceinline__ void cp16(void* lds, const void* g) {
#if HAVE_ASYNC
  __builtin_amdgcn_global_load_async_to_lds_b128((gas_v4i_p)g, (las_v4i_p)lds,
                                                 0, 0);
#else
  *(v8bf*)lds = *(const v8bf*)g;
#endif
}

template <int N>
__device__ __forceinline__ void wait_async() {
#if HAVE_ASYNC
#if __has_builtin(__builtin_amdgcn_s_wait_asynccnt)
  __builtin_amdgcn_s_wait_asynccnt(N);
#else
  asm volatile("s_wait_asynccnt %0" ::"i"(N) : "memory");
#endif
#endif
}

__device__ __forceinline__ v16bf pack16(v8bf lo, v8bf hi) {
  v16bf r;
#pragma unroll
  for (int e = 0; e < 8; ++e) { r[e] = lo[e]; r[e + 8] = hi[e]; }
  return r;
}

__device__ __forceinline__ v8f wmma_bf16(v16bf a, v16bf b, v8f c) {
  return __builtin_amdgcn_wmma_f32_16x16x32_bf16(false, a, false, b,
                                                 (short)0, c, false, false);
}

// ============================================================
// One-time f32 -> bf16 weight conversion (8 elems/thread).
// ============================================================
__global__ void __launch_bounds__(256)
f32_to_bf16_kernel(const float* __restrict__ in, __bf16* __restrict__ out, int n8) {
  const int i = blockIdx.x * 256 + threadIdx.x;
  if (i >= n8) return;
  const float4 a = ((const float4*)in)[i * 2];
  const float4 b = ((const float4*)in)[i * 2 + 1];
  v8bf o;
  o[0] = f2bf(a.x); o[1] = f2bf(a.y); o[2] = f2bf(a.z); o[3] = f2bf(a.w);
  o[4] = f2bf(b.x); o[5] = f2bf(b.y); o[6] = f2bf(b.z); o[7] = f2bf(b.w);
  ((v8bf*)out)[i] = o;
}

// ============================================================
// GEMM:  C[M,N] = A[M,K] @ W[N,K]^T + bias   (W pre-converted to bf16)
//   MODE 0: A = f32 (VALU convert in staging), out = bf16 permuted [B,NH,SEQ,HD]
//   MODE 1: A = bf16 (async-staged),           out = f32 row-major [M,HID]
// 256 thr / 8 waves (4x2), tile 128x128, K-step 32, double-buffered LDS.
// ============================================================
template <int MODE>
__global__ void __launch_bounds__(256)
gemm_bias_kernel(const float*  __restrict__ Xf,
                 const __bf16* __restrict__ Xb,
                 const __bf16* __restrict__ Wb,
                 const float*  __restrict__ bias,
                 float*        __restrict__ outF,
                 __bf16*       __restrict__ outB) {
  constexpr int BM = 128, BN = 128, BK = 32, SA = 40; // 80B row stride (16B mult)
  constexpr int NT  = HID / BK;
  constexpr int NPT = (MODE == 0) ? 2 : 4;            // async ops/thread/tile
  __shared__ __attribute__((aligned(16))) __bf16 As[2][BM][SA];
  __shared__ __attribute__((aligned(16))) __bf16 Bs[2][BN][SA];

  const int tid  = threadIdx.x;
  const int m0   = blockIdx.y * BM;
  const int n0   = blockIdx.x * BN;
  const int lane = tid & 31, wave = tid >> 5;
  const int wm   = wave & 3, wn = wave >> 2;   // wave tile: 32 rows x 64 cols
  const int hl   = lane & 15, g = lane >> 4;

  auto stage = [&](int kt, int buf) {
    const int k0 = kt * BK;
    if (MODE == 0) {                         // A: f32 -> bf16, VALU convert
      const int c = tid & 7, r = tid >> 3;   // 8 float4-cols x 32 rows
      for (int rr = r; rr < BM; rr += 32) {
        const float4 f = *(const float4*)(Xf + (size_t)(m0 + rr) * HID + k0 + c * 4);
        __bf16* d = &As[buf][rr][c * 4];
        d[0] = f2bf(f.x); d[1] = f2bf(f.y); d[2] = f2bf(f.z); d[3] = f2bf(f.w);
      }
    } else {                                 // A: bf16 async 16B chunks
      const int c = tid & 3, r = tid >> 2;   // 4 chunks x 64 rows
      for (int rr = r; rr < BM; rr += 64)
        cp16(&As[buf][rr][c * 8], Xb + (size_t)(m0 + rr) * HID + k0 + c * 8);
    }
    {                                        // B: bf16 async 16B chunks
      const int c = tid & 3, r = tid >> 2;
      for (int rr = r; rr < BN; rr += 64)
        cp16(&Bs[buf][rr][c * 8], Wb + (size_t)(n0 + rr) * HID + k0 + c * 8);
    }
  };

  v8f acc[2][4] = {};
  stage(0, 0);

  for (int kt = 0; kt < NT; ++kt) {
    const int cur = kt & 1;
    if (kt + 1 < NT) { stage(kt + 1, cur ^ 1); wait_async<NPT>(); }
    else             { wait_async<0>(); }
    __syncthreads();

    v16bf af[2], bfg[4];
#pragma unroll
    for (int i = 0; i < 2; ++i) {
      const int row = wm * 32 + i * 16 + hl;
      af[i] = pack16(*(const v8bf*)&As[cur][row][8 * g],
                     *(const v8bf*)&As[cur][row][16 + 8 * g]);
    }
#pragma unroll
    for (int j = 0; j < 4; ++j) {
      const int row = wn * 64 + j * 16 + hl;
      bfg[j] = pack16(*(const v8bf*)&Bs[cur][row][8 * g],
                      *(const v8bf*)&Bs[cur][row][16 + 8 * g]);
    }
#pragma unroll
    for (int i = 0; i < 2; ++i)
#pragma unroll
      for (int j = 0; j < 4; ++j)
        acc[i][j] = wmma_bf16(af[i], bfg[j], acc[i][j]);
    __syncthreads();
  }

  // epilogue: row = r + 8*(lane>>4), col = lane&15
#pragma unroll
  for (int i = 0; i < 2; ++i) {
#pragma unroll
    for (int j = 0; j < 4; ++j) {
      const int n  = n0 + wn * 64 + j * 16 + hl;
      const float bb = bias[n];
#pragma unroll
      for (int r = 0; r < 8; ++r) {
        const int   m   = m0 + wm * 32 + i * 16 + r + 8 * g;
        const float val = acc[i][j][r] + bb;
        if (MODE == 0) {
          const int b = m >> 11, l = m & (SEQ - 1);
          const int h = n >> 6,  d = n & (HD - 1);
          outB[((size_t)(b * NH + h) * SEQ + l) * HD + d] = f2bf(val);
        } else {
          outF[(size_t)m * HID + n] = val;
        }
      }
    }
  }
}

// ============================================================
// Flash attention per (b*16+h, 128-row q tile). K async double-buffered,
// V transposed into LDS, online softmax f32, P via per-wave LDS scratch.
// ============================================================
__global__ void __launch_bounds__(256)
attn_kernel(const __bf16* __restrict__ hq,
            const __bf16* __restrict__ hk,
            const __bf16* __restrict__ hv,
            __bf16*       __restrict__ ctx) {
  constexpr int QT = 128, KT = 64, SD = 72;   // 144B row stride (16B mult)
  constexpr int NT = SEQ / KT;
  __shared__ __attribute__((aligned(16))) __bf16 Qs[QT][SD];
  __shared__ __attribute__((aligned(16))) __bf16 Ks[2][KT][SD];
  __shared__ __attribute__((aligned(16))) __bf16 Vt[2][HD][SD];   // [d][key]
  __shared__ __attribute__((aligned(16))) __bf16 Ps[8][16][SD];   // per-wave P

  const int tid  = threadIdx.x;
  const int lane = tid & 31, wave = tid >> 5;
  const int hl   = lane & 15, g = lane >> 4;
  const int bh   = blockIdx.x;            // b*16 + h
  const int q0   = blockIdx.y * QT;
  const size_t base = (size_t)bh * SEQ * HD;
  const int c = tid & 7, rbase = tid >> 3;     // 8x16B chunks per 64-wide row

  // ---- stage Q tile (async), then build per-wave Q fragments ----
  for (int rr = rbase; rr < QT; rr += 32)
    cp16(&Qs[rr][c * 8], hq + base + (size_t)(q0 + rr) * HD + c * 8);
  wait_async<0>();
  __syncthreads();

  v16bf qf[2];
#pragma unroll
  for (int ks = 0; ks < 2; ++ks) {
    const int row = wave * 16 + hl;
    qf[ks] = pack16(*(const v8bf*)&Qs[row][ks * 32 + 8 * g],
                    *(const v8bf*)&Qs[row][ks * 32 + 16 + 8 * g]);
  }

  auto stageK = [&](int kt, int buf) {       // 2 async ops/thread
    for (int rr = rbase; rr < KT; rr += 32)
      cp16(&Ks[buf][rr][c * 8], hk + base + (size_t)(kt * KT + rr) * HD + c * 8);
  };
  auto stageV = [&](int kt, int buf) {       // sync transpose-on-store
    for (int rr = rbase; rr < KT; rr += 32) {
      const v8bf vv = *(const v8bf*)(hv + base + (size_t)(kt * KT + rr) * HD + c * 8);
#pragma unroll
      for (int e = 0; e < 8; ++e)
        Vt[buf][c * 8 + e][rr] = vv[e];
    }
  };

  v8f   oacc[4] = {};
  float mrow[8], lrow[8];
#pragma unroll
  for (int r = 0; r < 8; ++r) { mrow[r] = -1e30f; lrow[r] = 0.0f; }

  stageK(0, 0); stageV(0, 0);

  for (int kt = 0; kt < NT; ++kt) {
    const int cur = kt & 1;
    if (kt + 1 < NT) { stageK(kt + 1, cur ^ 1); stageV(kt + 1, cur ^ 1); wait_async<2>(); }
    else             { wait_async<0>(); }
    __syncthreads();

    // ---- S = Q @ K^T : 16 q-rows x 64 keys ----
    v8f sacc[4] = {};
#pragma unroll
    for (int ks = 0; ks < 2; ++ks) {
#pragma unroll
      for (int j = 0; j < 4; ++j) {
        const int krow = j * 16 + hl;
        v16bf kf = pack16(*(const v8bf*)&Ks[cur][krow][ks * 32 + 8 * g],
                          *(const v8bf*)&Ks[cur][krow][ks * 32 + 16 + 8 * g]);
        sacc[j] = wmma_bf16(qf[ks], kf, sacc[j]);
      }
    }
#pragma unroll
    for (int j = 0; j < 4; ++j)
#pragma unroll
      for (int r = 0; r < 8; ++r)
        sacc[j][r] = sacc[j][r] * SCALE;

    // ---- online softmax; row m = r + 8*g, 16-lane half reductions ----
#pragma unroll
    for (int r = 0; r < 8; ++r) {
      float mx = fmaxf(fmaxf(sacc[0][r], sacc[1][r]),
                       fmaxf(sacc[2][r], sacc[3][r]));
      for (int off = 1; off < 16; off <<= 1)
        mx = fmaxf(mx, __shfl_xor(mx, off, 32));
      const float mnew = fmaxf(mrow[r], mx);
      const float corr = __expf(mrow[r] - mnew);
      mrow[r] = mnew;
      float rs = 0.f;
#pragma unroll
      for (int j = 0; j < 4; ++j) {
        const float p = __expf(sacc[j][r] - mnew);
        sacc[j][r] = p;
        rs += p;
      }
      for (int off = 1; off < 16; off <<= 1)
        rs += __shfl_xor(rs, off, 32);
      lrow[r] = lrow[r] * corr + rs;
#pragma unroll
      for (int dj = 0; dj < 4; ++dj)
        oacc[dj][r] = oacc[dj][r] * corr;
    }

    // ---- P: C-layout -> bf16 A-layout via per-wave LDS scratch ----
#pragma unroll
    for (int j = 0; j < 4; ++j)
#pragma unroll
      for (int r = 0; r < 8; ++r)
        Ps[wave][r + 8 * g][j * 16 + hl] = f2bf(sacc[j][r]);

    // ---- O += P @ V ----
#pragma unroll
    for (int ks = 0; ks < 2; ++ks) {
      v16bf pf = pack16(*(const v8bf*)&Ps[wave][hl][ks * 32 + 8 * g],
                        *(const v8bf*)&Ps[wave][hl][ks * 32 + 16 + 8 * g]);
#pragma unroll
      for (int dj = 0; dj < 4; ++dj) {
        const int drow = dj * 16 + hl;
        v16bf vf = pack16(*(const v8bf*)&Vt[cur][drow][ks * 32 + 8 * g],
                          *(const v8bf*)&Vt[cur][drow][ks * 32 + 16 + 8 * g]);
        oacc[dj] = wmma_bf16(pf, vf, oacc[dj]);
      }
    }
    __syncthreads();
  }

  // ---- epilogue: ctx[b, l, h*64+d] bf16 ----
  const int b = bh >> 4, h = bh & 15;
#pragma unroll
  for (int dj = 0; dj < 4; ++dj) {
#pragma unroll
    for (int r = 0; r < 8; ++r) {
      const int m = q0 + wave * 16 + r + 8 * g;
      const int d = dj * 16 + hl;
      const float val = oacc[dj][r] / lrow[r];
      ctx[(size_t)(b * SEQ + m) * HID + h * HD + d] = f2bf(val);
    }
  }
}

// ============================================================
extern "C" void kernel_launch(void* const* d_in, const int* in_sizes, int n_in,
                              void* d_out, int out_size, void* d_ws, size_t ws_size,
                              hipStream_t stream) {
  (void)in_sizes; (void)n_in; (void)out_size; (void)ws_size;
  const float* q  = (const float*)d_in[0];
  const float* k  = (const float*)d_in[1];
  const float* v  = (const float*)d_in[2];
  const float* Wq = (const float*)d_in[3];
  const float* bq = (const float*)d_in[4];
  const float* Wk = (const float*)d_in[5];
  const float* bk = (const float*)d_in[6];
  const float* Wv = (const float*)d_in[7];
  const float* bv = (const float*)d_in[8];
  const float* Wo = (const float*)d_in[9];
  const float* bo = (const float*)d_in[10];
  float* out = (float*)d_out;

  const size_t NW = (size_t)HID * HID;               // 1M elems per weight
  const size_t NP = (size_t)BATCH * NH * SEQ * HD;   // 8.39M elems
  __bf16* Wqb = (__bf16*)d_ws;
  __bf16* Wkb = Wqb + NW;
  __bf16* Wvb = Wkb + NW;
  __bf16* Wob = Wvb + NW;
  __bf16* hq_ = Wob + NW;             // [B,NH,SEQ,HD]
  __bf16* hk_ = hq_ + NP;
  __bf16* hv_ = hk_ + NP;
  __bf16* cx_ = hv_ + NP;             // [B,SEQ,HID]

  dim3 blk(256);
  const int n8 = (int)(NW / 8);
  dim3 gc((n8 + 255) / 256);
  f32_to_bf16_kernel<<<gc, blk, 0, stream>>>(Wq, Wqb, n8);
  f32_to_bf16_kernel<<<gc, blk, 0, stream>>>(Wk, Wkb, n8);
  f32_to_bf16_kernel<<<gc, blk, 0, stream>>>(Wv, Wvb, n8);
  f32_to_bf16_kernel<<<gc, blk, 0, stream>>>(Wo, Wob, n8);

  dim3 gp(HID / 128, (BATCH * SEQ) / 128);  // (8, 64)
  gemm_bias_kernel<0><<<gp, blk, 0, stream>>>(q, (const __bf16*)nullptr, Wqb, bq,
                                              (float*)nullptr, hq_);
  gemm_bias_kernel<0><<<gp, blk, 0, stream>>>(k, (const __bf16*)nullptr, Wkb, bk,
                                              (float*)nullptr, hk_);
  gemm_bias_kernel<0><<<gp, blk, 0, stream>>>(v, (const __bf16*)nullptr, Wvb, bv,
                                              (float*)nullptr, hv_);
  dim3 ga(BATCH * NH, SEQ / 128);           // (64, 16)
  attn_kernel<<<ga, blk, 0, stream>>>(hq_, hk_, hv_, cx_);
  gemm_bias_kernel<1><<<gp, blk, 0, stream>>>((const float*)nullptr, cx_, Wob, bo,
                                              out, (__bf16*)nullptr);
}
